// FASA_23201413333539
// MI455X (gfx1250) — matrix operand order repaired and checked by reading, not compiled, not run on hardware
//
#include <hip/hip_runtime.h>

typedef __attribute__((ext_vector_type(16))) _Float16 v16h;
typedef __attribute__((ext_vector_type(8)))  float    v8f;
typedef unsigned int u32x4 __attribute__((ext_vector_type(4)));
typedef int          i32x8 __attribute__((ext_vector_type(8)));
typedef int          i32x4 __attribute__((ext_vector_type(4)));

#define NB    4
#define CCH   128
#define HIMG  128
#define WIMG  128
#define HWQ   (HIMG*WIMG)     // 16384
#define NHEAD 4
#define DHD   32
#define HWKV  1024
#define NQBLK (HWQ/16)        // 1024 q-blocks per (b,head)
#define BNEPS 1e-5f

#if defined(__has_builtin)
#  if __has_builtin(__builtin_amdgcn_tensor_load_to_lds) && __has_builtin(__builtin_amdgcn_s_wait_tensorcnt)
#    define HAVE_TDM 1
#  endif
#endif
#if __has_include(<hip/amd_detail/amd_gfx1250_TDM.h>)
#  define TDM_6ARG 1
#endif

__device__ __forceinline__ float sigmoidf_(float x) { return 1.0f / (1.0f + __expf(-x)); }

// ---------------------------------------------------------------------------
// Pointwise (1x1) conv as WMMA GEMM:  Y[b][o][p] = sum_c W[o][c]*X[b][c][p] + bias[o]
// One block handles (batch, 64-pixel tile). The Cin x 64 f32 X macro-tile is
// DMA'd into LDS once by the Tensor Data Mover; 8 waves cover Cout/16 M-tiles.
// ---------------------------------------------------------------------------
__global__ __launch_bounds__(256)
void pw_conv_wmma(const float* __restrict__ X, const float* __restrict__ W,
                  const float* __restrict__ bias, float* __restrict__ Y,
                  int Cout, int Cin, int P, int nB) {
  __shared__ float xtile[CCH * 64];          // 32 KB (Cin == 128 for all calls)
  const int lane  = threadIdx.x & 31;
  const int wave  = threadIdx.x >> 5;
  const int ntCnt = P >> 6;
  const int mtCnt = Cout >> 4;
  const int b  = blockIdx.x / ntCnt;
  const int nt = blockIdx.x % ntCnt;

  const float* Xb = X + (size_t)b * Cin * P;
  float*       Yb = Y + (size_t)b * Cout * P;

  // ---- stage X tile (Cin rows x 64 pixels) into LDS ----
#if defined(HAVE_TDM)
  if (wave == 0) {
    unsigned long long gaddr = (unsigned long long)(const void*)(Xb + (size_t)nt * 64);
    unsigned ldsOff = (unsigned)(unsigned long long)(const void*)&xtile[0];
    u32x4 g0;
    g0[0] = 1u;                                   // count=1, user mode, no gather
    g0[1] = ldsOff;                               // lds_addr
    g0[2] = (unsigned)(gaddr & 0xffffffffull);    // global_addr[31:0]
    g0[3] = (unsigned)((gaddr >> 32) & 0x1ffffffull) | (2u << 30); // addr[56:32], type=2
    unsigned long long s0 = (unsigned long long)P; // tensor_dim0_stride (elements)
    i32x8 g1;
    g1[0] = (int)(2u << 16);                      // data_size=4B (code 2)
    g1[1] = (int)(((unsigned)P & 0xffffu) << 16); // tensor_dim0[15:0] << 16
    g1[2] = (int)((((unsigned)P >> 16) & 0xffffu) | (((unsigned)Cin & 0xffffu) << 16));
    g1[3] = (int)((((unsigned)Cin >> 16) & 0xffffu) | (64u << 16)); // tile_dim0 = 64
    g1[4] = (int)((unsigned)Cin & 0xffffu);       // tile_dim1 = Cin, tile_dim2 = 0
    g1[5] = (int)(unsigned)(s0 & 0xffffffffull);  // dim0_stride[31:0]
    g1[6] = (int)(unsigned)((s0 >> 32) & 0xffffull); // dim0_stride[47:32], dim1_stride=0
    g1[7] = 0;
    i32x4 z4 = {0, 0, 0, 0};
#if defined(TDM_6ARG)
    i32x8 z8 = {0, 0, 0, 0, 0, 0, 0, 0};
    __builtin_amdgcn_tensor_load_to_lds(g0, g1, z4, z4, z8, 0);
#else
    __builtin_amdgcn_tensor_load_to_lds(g0, g1, z4, z4, 0);
#endif
    __builtin_amdgcn_s_wait_tensorcnt(0);
  }
  __syncthreads();
#else
  for (int idx = threadIdx.x; idx < Cin * 64; idx += 256) {
    int row = idx >> 6, col = idx & 63;
    xtile[idx] = Xb[(size_t)row * P + nt * 64 + col];
  }
  __syncthreads();
#endif

  const int khalf = (lane < 16) ? 0 : 8;    // A-layout K half (ISA 7.12.2)
  const int kbase = (lane < 16) ? 0 : 16;   // unused; B fragments also use khalf
  const int ncol  = lane & 15;

  for (int mt = wave; mt < mtCnt; mt += 8) {
    const int m = mt * 16 + ncol;
    v8f acc0 = {}, acc1 = {}, acc2 = {}, acc3 = {};
    for (int kc = 0; kc < Cin; kc += 32) {
      // A fragment: weights 16x32
      v16h a;
#pragma unroll
      for (int j = 0; j < 8; ++j) {
        int kb = ((j < 4) ? 2 * j : 16 + 2 * (j - 4)) + khalf;
        a[2 * j]     = (_Float16)W[(size_t)m * Cin + kc + kb];
        a[2 * j + 1] = (_Float16)W[(size_t)m * Cin + kc + kb + 1];
      }
      // B fragments from LDS: X 32x16, four N-tiles
      v16h bf0, bf1, bf2, bf3;
#pragma unroll
      for (int j = 0; j < 8; ++j) {
        const float* r0 = &xtile[(kc + ((j < 4) ? 2 * j : 16 + 2 * (j - 4)) + khalf) * 64];
        const float* r1 = r0 + 64;
        bf0[2 * j] = (_Float16)r0[ncol];      bf0[2 * j + 1] = (_Float16)r1[ncol];
        bf1[2 * j] = (_Float16)r0[ncol + 16]; bf1[2 * j + 1] = (_Float16)r1[ncol + 16];
        bf2[2 * j] = (_Float16)r0[ncol + 32]; bf2[2 * j + 1] = (_Float16)r1[ncol + 32];
        bf3[2 * j] = (_Float16)r0[ncol + 48]; bf3[2 * j + 1] = (_Float16)r1[ncol + 48];
      }
      acc0 = __builtin_amdgcn_wmma_f32_16x16x32_f16(false, a, false, bf0, (short)0, acc0, false, false);
      acc1 = __builtin_amdgcn_wmma_f32_16x16x32_f16(false, a, false, bf1, (short)0, acc1, false, false);
      acc2 = __builtin_amdgcn_wmma_f32_16x16x32_f16(false, a, false, bf2, (short)0, acc2, false, false);
      acc3 = __builtin_amdgcn_wmma_f32_16x16x32_f16(false, a, false, bf3, (short)0, acc3, false, false);
    }
    // D layout: VGPR r -> M = r + 8*(lane>=16), N = lane&15
    const int mbase = mt * 16 + ((lane < 16) ? 0 : 8);
    const int n0 = nt * 64 + ncol;
#pragma unroll
    for (int r = 0; r < 8; ++r) {
      const int mo = mbase + r;
      const float bv = bias[mo];
      float* yr = Yb + (size_t)mo * P;
      yr[n0]      = acc0[r] + bv;
      yr[n0 + 16] = acc1[r] + bv;
      yr[n0 + 32] = acc2[r] + bv;
      yr[n0 + 48] = acc3[r] + bv;
    }
  }
}

// ---------------------------------------------------------------------------
// Depthwise 5x5 conv (pad 2) + optional inference BN.
// ---------------------------------------------------------------------------
__global__ __launch_bounds__(256)
void dw_conv5(const float* __restrict__ in, const float* __restrict__ w,
              const float* __restrict__ bias,
              const float* __restrict__ bn_g, const float* __restrict__ bn_b,
              const float* __restrict__ bn_m, const float* __restrict__ bn_v,
              float* __restrict__ out, int nB, int ch,
              int Hin, int Win, int Hout, int Wout, int stride, int has_bn) {
  long long i = (long long)blockIdx.x * blockDim.x + threadIdx.x;
  long long total = (long long)nB * ch * Hout * Wout;
  if (i >= total) return;
  int ox = i % Wout;
  int oy = (i / Wout) % Hout;
  int c  = (i / ((long long)Wout * Hout)) % ch;
  int b  = i / ((long long)Wout * Hout * ch);
  const float* ip = in + ((size_t)b * ch + c) * Hin * Win;
  const float* wp = w + (size_t)c * 25;
  float acc = 0.0f;
#pragma unroll
  for (int ky = 0; ky < 5; ++ky) {
    int iy = oy * stride - 2 + ky;
    if (iy < 0 || iy >= Hin) continue;
#pragma unroll
    for (int kx = 0; kx < 5; ++kx) {
      int ix = ox * stride - 2 + kx;
      if (ix < 0 || ix >= Win) continue;
      acc += wp[ky * 5 + kx] * ip[(size_t)iy * Win + ix];
    }
  }
  acc += bias[c];
  if (has_bn) {
    float sc = bn_g[c] * rsqrtf(bn_v[c] + BNEPS);
    acc = (acc - bn_m[c]) * sc + bn_b[c];
  }
  out[i] = acc;
}

// ---------------------------------------------------------------------------
// (C,P) f32 channel-major -> (P,C) f16 row-major, per batch.
// ---------------------------------------------------------------------------
__global__ __launch_bounds__(256)
void transpose_f16(const float* __restrict__ in, _Float16* __restrict__ out,
                   int nB, int Cc, int P) {
  long long i = (long long)blockIdx.x * blockDim.x + threadIdx.x;
  long long total = (long long)nB * Cc * P;
  if (i >= total) return;
  int c = i % Cc;
  int p = (i / Cc) % P;
  int b = i / ((long long)Cc * P);
  out[i] = (_Float16)in[((size_t)b * Cc + c) * P + p];
}

// ---------------------------------------------------------------------------
// Flash-style attention: one wave per 16-row q block of one (b, head).
// qh: (B, HWQ, 128) f16 ; kvh: (B, HWKV, 256) f16 (k = ch 0..127, v = ch 128..255)
// gf: (B, 128, HWQ) f32
// ---------------------------------------------------------------------------
__global__ __launch_bounds__(256)
void fasa_attention(const _Float16* __restrict__ qh, const _Float16* __restrict__ kvh,
                    float* __restrict__ gf) {
  __shared__ _Float16 plds[8][16 * 32];   // per-wave P staging (D-layout -> A-layout)
  const int lane = threadIdx.x & 31;
  const int wave = threadIdx.x >> 5;
  int wid = blockIdx.x * 8 + wave;
  const int b   = wid / (NHEAD * NQBLK);
  int rem       = wid % (NHEAD * NQBLK);
  const int h   = rem / NQBLK;
  const int qb  = rem % NQBLK;

  const _Float16* qbase  = qh  + (size_t)b * HWQ * CCH;
  const _Float16* kvbase = kvh + (size_t)b * HWKV * (2 * CCH);

  const int khalf = (lane < 16) ? 0 : 8;    // A-layout half
  const int kgrp  = (lane < 16) ? 0 : 16;   // unused; K/V fragments also use khalf
  const int col   = lane & 15;
  const int hlf   = (lane < 16) ? 0 : 1;
  const float scale = 0.17677669529663687f; // 32^-0.5

  // Q fragment (16x32, dh contiguous in qh)
  const int mq = qb * 16 + col;
  v16h aQ;
#pragma unroll
  for (int j = 0; j < 8; ++j) {
    int db = ((j < 4) ? 2 * j : 16 + 2 * (j - 4)) + khalf;
    aQ[2 * j]     = qbase[(size_t)mq * CCH + h * DHD + db];
    aQ[2 * j + 1] = qbase[(size_t)mq * CCH + h * DHD + db + 1];
  }

  float mrow[8], lrow[8];
#pragma unroll
  for (int r = 0; r < 8; ++r) { mrow[r] = -1e30f; lrow[r] = 0.0f; }
  v8f o0 = {}, o1 = {};

  for (int kt = 0; kt < HWKV; kt += 32) {
    // K fragments for two groups of 16 keys
    v16h bK0, bK1;
    {
      const _Float16* kr0 = kvbase + (size_t)(kt + col) * (2 * CCH) + h * DHD + khalf;
      const _Float16* kr1 = kvbase + (size_t)(kt + 16 + col) * (2 * CCH) + h * DHD + khalf;
#pragma unroll
      for (int j = 0; j < 8; ++j) {
        int kb = (j < 4) ? 2 * j : 16 + 2 * (j - 4);
        bK0[2 * j] = kr0[kb]; bK0[2 * j + 1] = kr0[kb + 1];
        bK1[2 * j] = kr1[kb]; bK1[2 * j + 1] = kr1[kb + 1];
      }
    }
    v8f S0 = {}, S1 = {};
    S0 = __builtin_amdgcn_wmma_f32_16x16x32_f16(false, aQ, false, bK0, (short)0, S0, false, false);
    S1 = __builtin_amdgcn_wmma_f32_16x16x32_f16(false, aQ, false, bK1, (short)0, S1, false, false);

    // online softmax: rows live in VGPR r across 16-lane halves
    float p0[8], p1[8], alpha[8];
#pragma unroll
    for (int r = 0; r < 8; ++r) {
      float s0 = S0[r] * scale, s1 = S1[r] * scale;
      float rmax = fmaxf(s0, s1);
      rmax = fmaxf(rmax, __shfl_xor(rmax, 1));
      rmax = fmaxf(rmax, __shfl_xor(rmax, 2));
      rmax = fmaxf(rmax, __shfl_xor(rmax, 4));
      rmax = fmaxf(rmax, __shfl_xor(rmax, 8));
      float mnew = fmaxf(mrow[r], rmax);
      float al   = __expf(mrow[r] - mnew);
      float e0   = __expf(s0 - mnew);
      float e1   = __expf(s1 - mnew);
      float rs = e0 + e1;
      rs += __shfl_xor(rs, 1);
      rs += __shfl_xor(rs, 2);
      rs += __shfl_xor(rs, 4);
      rs += __shfl_xor(rs, 8);
      lrow[r] = lrow[r] * al + rs;
      mrow[r] = mnew;
      alpha[r] = al; p0[r] = e0; p1[r] = e1;
    }
#pragma unroll
    for (int r = 0; r < 8; ++r) { o0[r] *= alpha[r]; o1[r] *= alpha[r]; }

    // stage P (D layout) into LDS, reload as A fragment
#pragma unroll
    for (int r = 0; r < 8; ++r) {
      int row = hlf * 8 + r;
      plds[wave][row * 32 + col]      = (_Float16)p0[r];
      plds[wave][row * 32 + 16 + col] = (_Float16)p1[r];
    }
    asm volatile("s_wait_dscnt 0x0" ::: "memory");
    v16h aP;
#pragma unroll
    for (int j = 0; j < 8; ++j) {
      int kb = ((j < 4) ? 2 * j : 16 + 2 * (j - 4)) + khalf;
      aP[2 * j]     = plds[wave][col * 32 + kb];
      aP[2 * j + 1] = plds[wave][col * 32 + kb + 1];
    }
    // V fragments: B[k=key_local][n=d], d fixed per lane, keys strided
    v16h bV0, bV1;
#pragma unroll
    for (int j = 0; j < 8; ++j) {
      int keyl = kt + ((j < 4) ? 2 * j : 16 + 2 * (j - 4)) + khalf;
      const _Float16* vr0 = kvbase + (size_t)keyl * (2 * CCH) + CCH + h * DHD;
      const _Float16* vr1 = vr0 + 2 * CCH;
      bV0[2 * j] = vr0[col];      bV0[2 * j + 1] = vr1[col];
      bV1[2 * j] = vr0[col + 16]; bV1[2 * j + 1] = vr1[col + 16];
    }
    o0 = __builtin_amdgcn_wmma_f32_16x16x32_f16(false, aP, false, bV0, (short)0, o0, false, false);
    o1 = __builtin_amdgcn_wmma_f32_16x16x32_f16(false, aP, false, bV1, (short)0, o1, false, false);
  }

  // normalize + store to gf (NCHW f32)
#pragma unroll
  for (int r = 0; r < 8; ++r) {
    float inv = 1.0f / lrow[r];
    int p = qb * 16 + hlf * 8 + r;
    gf[((size_t)b * CCH + h * DHD + col) * HWQ + p]      = o0[r] * inv;
    gf[((size_t)b * CCH + h * DHD + col + 16) * HWQ + p] = o1[r] * inv;
  }
}

// ---------------------------------------------------------------------------
// lf = lf*sigmoid(lf)*sigmoid(gf)*gf  (in-place on lf)
// ---------------------------------------------------------------------------
__global__ __launch_bounds__(256)
void gate_fuse(float* __restrict__ lf, const float* __restrict__ gf, long long n) {
  long long i = (long long)blockIdx.x * blockDim.x + threadIdx.x;
  if (i >= n) return;
  float l = lf[i], g = gf[i];
  lf[i] = l * sigmoidf_(l) * sigmoidf_(g) * g;
}

// ---------------------------------------------------------------------------
extern "C" void kernel_launch(void* const* d_in, const int* in_sizes, int n_in,
                              void* d_out, int out_size, void* d_ws, size_t ws_size,
                              hipStream_t stream) {
  const float* x       = (const float*)d_in[0];
  const float* q_w     = (const float*)d_in[1];
  const float* q_b     = (const float*)d_in[2];
  const float* kv_w    = (const float*)d_in[3];
  const float* kv_b    = (const float*)d_in[4];
  const float* local_w = (const float*)d_in[5];
  const float* local_b = (const float*)d_in[6];
  const float* mixer_w = (const float*)d_in[7];
  const float* mixer_b = (const float*)d_in[8];
  const float* p0_w    = (const float*)d_in[9];
  const float* p0_b    = (const float*)d_in[10];
  const float* bn0_g   = (const float*)d_in[11];
  const float* bn0_b   = (const float*)d_in[12];
  const float* bn0_m   = (const float*)d_in[13];
  const float* bn0_v   = (const float*)d_in[14];
  const float* lin0_w  = (const float*)d_in[15];
  const float* lin0_b  = (const float*)d_in[16];
  const float* p1_w    = (const float*)d_in[17];
  const float* p1_b    = (const float*)d_in[18];
  const float* bn1_g   = (const float*)d_in[19];
  const float* bn1_b   = (const float*)d_in[20];
  const float* bn1_m   = (const float*)d_in[21];
  const float* bn1_v   = (const float*)d_in[22];

  char* ws = (char*)d_ws;
  size_t off = 0;
  auto take = [&](size_t bytes) -> char* {
    char* p = ws + off;
    off = (off + bytes + 255) & ~(size_t)255;
    return p;
  };
  float*    q_local = (float*)take((size_t)NB * CCH * HWQ * 4);
  float*    pA      = (float*)take((size_t)NB * CCH * 64 * 64 * 4);
  float*    pB      = (float*)take((size_t)NB * CCH * 64 * 64 * 4);
  float*    p2      = (float*)take((size_t)NB * CCH * HWKV * 4);
  float*    kvb     = (float*)take((size_t)NB * 2 * CCH * HWKV * 4);
  _Float16* qh      = (_Float16*)take((size_t)NB * HWQ * CCH * 2);
  _Float16* kvh     = (_Float16*)take((size_t)NB * HWKV * 2 * CCH * 2);
  float*    gf      = (float*)take((size_t)NB * CCH * HWQ * 4);
  float*    lf      = (float*)take((size_t)NB * CCH * HWQ * 4);
  (void)ws_size; (void)in_sizes; (void)n_in; (void)out_size;

  // 1) q_local = 1x1 conv (WMMA GEMM, TDM-staged X)
  pw_conv_wmma<<<NB * (HWQ / 64), 256, 0, stream>>>(x, q_w, q_b, q_local, CCH, CCH, HWQ, NB);
  // 2) pool: dw5x5 s2 + bn0
  {
    long long tot = (long long)NB * CCH * 64 * 64;
    dw_conv5<<<(int)((tot + 255) / 256), 256, 0, stream>>>(
        x, p0_w, p0_b, bn0_g, bn0_b, bn0_m, bn0_v, pA,
        NB, CCH, HIMG, WIMG, 64, 64, 2, 1);
  }
  // 3) lin0 1x1 conv
  pw_conv_wmma<<<NB * (64 * 64 / 64), 256, 0, stream>>>(pA, lin0_w, lin0_b, pB, CCH, CCH, 64 * 64, NB);
  // 4) dw5x5 s2 + bn1 -> p2 (B,128,32,32)
  {
    long long tot = (long long)NB * CCH * 32 * 32;
    dw_conv5<<<(int)((tot + 255) / 256), 256, 0, stream>>>(
        pB, p1_w, p1_b, bn1_g, bn1_b, bn1_m, bn1_v, p2,
        NB, CCH, 64, 64, 32, 32, 2, 1);
  }
  // 5) kv = 1x1 conv 128->256
  pw_conv_wmma<<<NB * (HWKV / 64), 256, 0, stream>>>(p2, kv_w, kv_b, kvb, 2 * CCH, CCH, HWKV, NB);
  // 6) f16 transposes for fragment-friendly layouts
  {
    long long tot = (long long)NB * CCH * HWQ;
    transpose_f16<<<(int)((tot + 255) / 256), 256, 0, stream>>>(q_local, qh, NB, CCH, HWQ);
    long long tot2 = (long long)NB * 2 * CCH * HWKV;
    transpose_f16<<<(int)((tot2 + 255) / 256), 256, 0, stream>>>(kvb, kvh, NB, 2 * CCH, HWKV);
  }
  // 7) flash attention -> gf
  fasa_attention<<<(NB * NHEAD * NQBLK) / 8, 256, 0, stream>>>(qh, kvh, gf);
  // 8) lf = dw5x5(q_local) stride 1, no bn
  {
    long long tot = (long long)NB * CCH * HWQ;
    dw_conv5<<<(int)((tot + 255) / 256), 256, 0, stream>>>(
        q_local, local_w, local_b, nullptr, nullptr, nullptr, nullptr, lf,
        NB, CCH, HIMG, WIMG, HIMG, WIMG, 1, 0);
  }
  // 9) gated fusion (in place on lf)
  {
    long long tot = (long long)NB * CCH * HWQ;
    gate_fuse<<<(int)((tot + 255) / 256), 256, 0, stream>>>(lf, gf, tot);
  }
  // 10) mixer 1x1 conv -> d_out
  pw_conv_wmma<<<NB * (HWQ / 64), 256, 0, stream>>>(lf, mixer_w, mixer_b, (float*)d_out, CCH, CCH, HWQ, NB);
}
